// ArmaNetwork_83726092468413
// MI455X (gfx1250) — matrix operand order repaired and checked by reading, not compiled, not run on hardware
//
#include <hip/hip_runtime.h>

typedef __attribute__((ext_vector_type(16))) _Float16 v16h;
typedef __attribute__((ext_vector_type(8)))  float    v8f;

#define N_NODES 50000
#define N_EDGES 800000

// ---------------------------------------------------------------- utilities
__global__ void zero_f32x4_kernel(float4* __restrict__ p, int n4) {
  int i = blockIdx.x * blockDim.x + threadIdx.x;
  if (i < n4) p[i] = make_float4(0.f, 0.f, 0.f, 0.f);
}

__global__ void zero_f32_kernel(float* __restrict__ p, int n) {
  int i = blockIdx.x * blockDim.x + threadIdx.x;
  if (i < n) p[i] = 0.0f;
}

__global__ void cvt_f32_to_f16_kernel(const float* __restrict__ s,
                                      _Float16* __restrict__ d, int n) {
  int i = blockIdx.x * blockDim.x + threadIdx.x;
  if (i < n) d[i] = (_Float16)s[i];
}

// ---------------------------------------------------------------- gcn_norm
__global__ void deg_kernel(const int* __restrict__ col, float* __restrict__ deg, int e) {
  int i = blockIdx.x * blockDim.x + threadIdx.x;
  if (i < e) atomicAdd(&deg[col[i]], 1.0f);
}

__global__ void dinv_kernel(const float* __restrict__ deg, float* __restrict__ dinv, int n) {
  int i = blockIdx.x * blockDim.x + threadIdx.x;
  if (i < n) { float d = deg[i]; dinv[i] = (d > 0.0f) ? rsqrtf(d) : 0.0f; }
}

__global__ void norm_kernel(const int* __restrict__ row, const int* __restrict__ col,
                            const float* __restrict__ dinv, float* __restrict__ nrm, int e) {
  int i = blockIdx.x * blockDim.x + threadIdx.x;
  if (i < e) nrm[i] = dinv[row[i]] * dinv[col[i]];
}

// ------------------------------------------------------------- weight packs
// W1cat^T : [256 n][128 k] half.  n<128 -> init(k=n/64,o=n%64), else root.
__global__ void pack_w1_kernel(const float* __restrict__ w_init,
                               const float* __restrict__ w_root,
                               _Float16* __restrict__ dst) {
  int i = blockIdx.x * blockDim.x + threadIdx.x;      // 256*128
  if (i >= 256 * 128) return;
  int n = i >> 7, f = i & 127;
  float v;
  if (n < 128) { int k = n >> 6, o = n & 63; v = w_init[k * 8192 + f * 64 + o]; }
  else { int m = n - 128; int k = m >> 6, o = m & 63; v = w_root[k * 8192 + f * 64 + o]; }
  dst[n * 128 + f] = (_Float16)v;
}

// per-stack 64x64 mid weight, transposed: dst[k][o*64 + i] = w[k][i][o]
__global__ void pack_wmid_kernel(const float* __restrict__ w, _Float16* __restrict__ dst) {
  int i = blockIdx.x * blockDim.x + threadIdx.x;      // 2*64*64
  if (i >= 8192) return;
  int k = i >> 12, r = i & 4095, o = r >> 6, ii = r & 63;
  dst[k * 4096 + o * 64 + ii] = (_Float16)w[k * 4096 + ii * 64 + o];
}

// W2cat^T : [48 n][64 k] half (rows 40..47 zero padding)
__global__ void pack_w2_kernel(const float* __restrict__ w_init,
                               const float* __restrict__ w_root,
                               _Float16* __restrict__ dst) {
  int i = blockIdx.x * blockDim.x + threadIdx.x;      // 48*64
  if (i >= 48 * 64) return;
  int n = i >> 6, f = i & 63;
  float v = 0.0f;
  if (n < 20)      { int k = n / 10, o = n % 10; v = w_init[k * 640 + f * 10 + o]; }
  else if (n < 40) { int m = n - 20; int k = m / 10, o = m % 10; v = w_root[k * 640 + f * 10 + o]; }
  dst[n * 64 + f] = (_Float16)v;
}

// -------------------------------------------------------------- WMMA GEMM
// C[M x N] = A[M x K] * B[K x N], A row-major f16, B pre-transposed
// (BT[N][K], n-major) f16, C row-major f32.  One wave per 16x16 tile,
// K-loop in chunks of 32 using v_wmma_f32_16x16x32_f16.
//   A 16x32 f16 lane layout: lane -> M = lane&15; VGPR r -> K = 2(r&3)+8g+16(r>>2)
//   B 32x16 f16 lane layout: lane -> N = lane&15; VGPR r -> K = 2(r&3)+8g+16(r>>2)
//   C/D f32 layout: lane -> N = lane&15; VGPR r -> M = r+8g
// Each 4-VGPR fragment group is K-contiguous -> compiler emits global_load_b128.
__global__ __launch_bounds__(256) void gemm_wmma_f16(
    const _Float16* __restrict__ A, int lda,
    const _Float16* __restrict__ BT, int ldb,
    float* __restrict__ C, int ldc,
    int mTiles, int nTiles, int kDim) {
  int wave = threadIdx.x >> 5;
  int lane = threadIdx.x & 31;
  int tile = blockIdx.x * 8 + wave;
  if (tile >= mTiles * nTiles) return;                // whole wave exits: EXEC all-1 for WMMA
  int tm = tile / nTiles, tn = tile % nTiles;
  int g  = lane >> 4;                                 // lane group (0/1)
  int l  = lane & 15;
  const _Float16* Arow = A  + (size_t)(tm * 16 + l) * lda;
  const _Float16* Bcol = BT + (size_t)(tn * 16 + l) * ldb;
  v8f acc = {};
  for (int kc = 0; kc < kDim; kc += 32) {
    union { v16h v; unsigned u[8]; } a, b;
#pragma unroll
    for (int r = 0; r < 8; ++r) {
      int ka = kc + 2 * (r & 3) + 8 * g + 16 * (r >> 2);
      int kb = kc + 2 * (r & 3) + 8 * g + 16 * (r >> 2);
      a.u[r] = *reinterpret_cast<const unsigned*>(Arow + ka);
      b.u[r] = *reinterpret_cast<const unsigned*>(Bcol + kb);
    }
    acc = __builtin_amdgcn_wmma_f32_16x16x32_f16(false, a.v, false, b.v,
                                                 (short)0, acc, false, false);
  }
#pragma unroll
  for (int r = 0; r < 8; ++r) {
    int rowi = tm * 16 + r + 8 * g;
    C[(size_t)rowi * ldc + tn * 16 + l] = acc[r];
  }
}

// ----------------------------------------------------- edge gather/scatter
// one wave per edge: dst[col*F + f] += src[row*sstride + f] * norm[e]
// F % 4 == 0 and all row strides keep 16B alignment -> float4 gathers.
__global__ __launch_bounds__(256) void prop_scatter_kernel(
    const float* __restrict__ src, int sstride,
    const float* __restrict__ nrm,
    const int* __restrict__ row, const int* __restrict__ col,
    float* __restrict__ dst, int F, int e) {
  int wid = (int)((blockIdx.x * (unsigned)blockDim.x + threadIdx.x) >> 5);
  int lane = threadIdx.x & 31;
  if (wid >= e) return;
  int r = row[wid], c = col[wid];
  float w = nrm[wid];
  const float* s = src + (size_t)r * sstride;
  float* d = dst + (size_t)c * F;
  __builtin_prefetch(s, 0, 1);                         // global_prefetch_b8
  for (int f4 = lane * 4; f4 < F; f4 += 128) {
    float4 v = *reinterpret_cast<const float4*>(s + f4);
    atomicAdd(d + f4 + 0, v.x * w);
    atomicAdd(d + f4 + 1, v.y * w);
    atomicAdd(d + f4 + 2, v.z * w);
    atomicAdd(d + f4 + 3, v.w * w);
  }
}

// ----------------------------------------------- layer-1 pointwise kernels
// t=0: out = relu(agg + root + bias) -> half (GEMM input for t=1)
__global__ void add_root_relu1_kernel(const float* __restrict__ tmp,
                                      const float* __restrict__ y1,
                                      const float* __restrict__ b1,
                                      _Float16* __restrict__ outh, int n) {
  int i = blockIdx.x * blockDim.x + threadIdx.x;       // N*128
  if (i >= n) return;
  int nd = i >> 7, f = i & 127;
  float v = tmp[i] + y1[(size_t)nd * 256 + 128 + f] + b1[f];
  outh[i] = (_Float16)fmaxf(v, 0.0f);
}

// t=1 finish: h = relu(mean_k(agg + root + bias)) -> half [N,64]
__global__ void final_l1_kernel(const float* __restrict__ tmp,
                                const float* __restrict__ y1,
                                const float* __restrict__ b1,
                                _Float16* __restrict__ hh, int n) {
  int i = blockIdx.x * blockDim.x + threadIdx.x;       // N*64
  if (i >= n) return;
  int nd = i >> 6, o = i & 63;
  float a = tmp[nd * 128 + o]      + y1[(size_t)nd * 256 + 128 + o] + b1[o];
  float b = tmp[nd * 128 + 64 + o] + y1[(size_t)nd * 256 + 192 + o] + b1[64 + o];
  hh[i] = (_Float16)fmaxf(0.5f * (a + b), 0.0f);
}

// ----------------------------------------------- layer-2 pointwise kernels
__global__ void add_root_relu2_kernel(const float* __restrict__ tmp,
                                      const float* __restrict__ y2,
                                      const float* __restrict__ b2,
                                      float* __restrict__ g1, int n) {
  int i = blockIdx.x * blockDim.x + threadIdx.x;       // N*20
  if (i >= n) return;
  int nd = i / 20, f = i % 20;
  float v = tmp[i] + y2[(size_t)nd * 48 + 20 + f] + b2[f];
  g1[i] = fmaxf(v, 0.0f);
}

// tiny per-stack 10x10 GEMM, weights staged in LDS
__global__ __launch_bounds__(256) void mid_gemm_w2_kernel(
    const float* __restrict__ X, const float* __restrict__ W,
    float* __restrict__ Y, int n) {
  __shared__ float wsh[200];
  for (int i = threadIdx.x; i < 200; i += blockDim.x) wsh[i] = W[i];
  __syncthreads();
  int idx = blockIdx.x * blockDim.x + threadIdx.x;     // N*20
  if (idx >= n) return;
  int nd = idx / 20, rem = idx % 20;
  int k = rem / 10, o = rem % 10;
  const float* xr = X + (size_t)nd * 20 + k * 10;
  const float* wk = wsh + k * 100;
  float acc = 0.0f;
#pragma unroll
  for (int i = 0; i < 10; ++i) acc += xr[i] * wk[i * 10 + o];
  Y[idx] = acc;
}

// final: mean over stacks, relu, log_softmax over 10 classes
__global__ void final_out_kernel(const float* __restrict__ tmp,
                                 const float* __restrict__ y2,
                                 const float* __restrict__ b2,
                                 float* __restrict__ out, int nNodes) {
  int n = blockIdx.x * blockDim.x + threadIdx.x;
  if (n >= nNodes) return;
  const float* t  = tmp + (size_t)n * 20;
  const float* rt = y2 + (size_t)n * 48 + 20;
  float v[10];
#pragma unroll
  for (int c = 0; c < 10; ++c) {
    float a = t[c]      + rt[c]      + b2[c];
    float b = t[10 + c] + rt[10 + c] + b2[10 + c];
    v[c] = fmaxf(0.5f * (a + b), 0.0f);
  }
  float mx = v[0];
#pragma unroll
  for (int c = 1; c < 10; ++c) mx = fmaxf(mx, v[c]);
  float s = 0.0f;
#pragma unroll
  for (int c = 0; c < 10; ++c) s += expf(v[c] - mx);
  float ls = logf(s);
#pragma unroll
  for (int c = 0; c < 10; ++c) out[(size_t)n * 10 + c] = v[c] - mx - ls;
}

// ---------------------------------------------------------------- launcher
static inline unsigned cdiv(unsigned a, unsigned b) { return (a + b - 1) / b; }

extern "C" void kernel_launch(void* const* d_in, const int* in_sizes, int n_in,
                              void* d_out, int out_size, void* d_ws, size_t ws_size,
                              hipStream_t stream) {
  (void)in_sizes; (void)n_in; (void)out_size; (void)ws_size;
  const int N = N_NODES, E = N_EDGES;

  const float* x       = (const float*)d_in[0];
  const int*   ei      = (const int*)d_in[1];
  const float* c1_init = (const float*)d_in[2];
  const float* c1_w    = (const float*)d_in[3];
  const float* c1_root = (const float*)d_in[4];
  const float* c1_bias = (const float*)d_in[5];
  const float* c2_init = (const float*)d_in[6];
  const float* c2_w    = (const float*)d_in[7];
  const float* c2_root = (const float*)d_in[8];
  const float* c2_bias = (const float*)d_in[9];
  const int* row = ei;
  const int* col = ei + E;
  float* out = (float*)d_out;

  char* ws = (char*)d_ws;
  size_t off = 0;
  auto alloc = [&](size_t bytes) { size_t o = off; off += (bytes + 255) & ~(size_t)255; return o; };
  float*    deg  = (float*)   (ws + alloc((size_t)N * 4));
  float*    dinv = (float*)   (ws + alloc((size_t)N * 4));
  float*    nrm  = (float*)   (ws + alloc((size_t)E * 4));
  _Float16* w1h  = (_Float16*)(ws + alloc((size_t)256 * 128 * 2));
  _Float16* wm1h = (_Float16*)(ws + alloc((size_t)8192 * 2));
  _Float16* w2h  = (_Float16*)(ws + alloc((size_t)48 * 64 * 2));
  _Float16* xh   = (_Float16*)(ws + alloc((size_t)N * 128 * 2));  // later reused as z2 [N,20] f32
  float*    y1   = (float*)   (ws + alloc((size_t)N * 256 * 4));  // later reused as Y2 [N,48]
  float*    tmp  = (float*)   (ws + alloc((size_t)N * 128 * 4));
  _Float16* outh = (_Float16*)(ws + alloc((size_t)N * 128 * 2));
  float*    z    = (float*)   (ws + alloc((size_t)N * 128 * 4));  // later reused as g1 [N,20]
  float* y2 = y1;
  float* z2 = (float*)xh;
  float* g1 = z;

  // ---- gcn_norm
  zero_f32_kernel<<<cdiv(N, 256), 256, 0, stream>>>(deg, N);
  deg_kernel<<<cdiv(E, 256), 256, 0, stream>>>(col, deg, E);
  dinv_kernel<<<cdiv(N, 256), 256, 0, stream>>>(deg, dinv, N);
  norm_kernel<<<cdiv(E, 256), 256, 0, stream>>>(row, col, dinv, nrm, E);

  // ---- conversions / weight packing
  cvt_f32_to_f16_kernel<<<cdiv(N * 128, 256), 256, 0, stream>>>(x, xh, N * 128);
  pack_w1_kernel  <<<cdiv(256 * 128, 256), 256, 0, stream>>>(c1_init, c1_root, w1h);
  pack_wmid_kernel<<<cdiv(8192, 256), 256, 0, stream>>>(c1_w, wm1h);
  pack_w2_kernel  <<<cdiv(48 * 64, 256), 256, 0, stream>>>(c2_init, c2_root, w2h);

  const int mT = N / 16;  // 3125 exactly
  // ---- layer 1: Y1[N,256] = X[N,128] * [init|root]  (WMMA)
  gemm_wmma_f16<<<cdiv(mT * 16, 8), 256, 0, stream>>>(xh, 128, w1h, 128, y1, 256, mT, 16, 128);

  // t=0 propagate + root + relu
  zero_f32x4_kernel<<<cdiv(N * 32, 256), 256, 0, stream>>>((float4*)tmp, N * 32);
  prop_scatter_kernel<<<cdiv(E * 32, 256), 256, 0, stream>>>(y1, 256, nrm, row, col, tmp, 128, E);
  add_root_relu1_kernel<<<cdiv(N * 128, 256), 256, 0, stream>>>(tmp, y1, c1_bias, outh, N * 128);

  // t=1: out @ w (64x64 per stack, WMMA), propagate, mean+relu -> h (half)
  for (int k = 0; k < 2; ++k)
    gemm_wmma_f16<<<cdiv(mT * 4, 8), 256, 0, stream>>>(outh + k * 64, 128, wm1h + k * 4096, 64,
                                                       z + k * 64, 128, mT, 4, 64);
  zero_f32x4_kernel<<<cdiv(N * 32, 256), 256, 0, stream>>>((float4*)tmp, N * 32);
  prop_scatter_kernel<<<cdiv(E * 32, 256), 256, 0, stream>>>(z, 128, nrm, row, col, tmp, 128, E);
  final_l1_kernel<<<cdiv(N * 64, 256), 256, 0, stream>>>(tmp, y1, c1_bias, outh, N * 64);

  // ---- layer 2: Y2[N,48] = h[N,64] * [init2|root2|pad]  (WMMA)
  gemm_wmma_f16<<<cdiv(mT * 3, 8), 256, 0, stream>>>(outh, 64, w2h, 64, y2, 48, mT, 3, 64);

  // t=0 propagate + root + relu (F=20)
  zero_f32x4_kernel<<<cdiv(N * 5, 256), 256, 0, stream>>>((float4*)tmp, N * 5);
  prop_scatter_kernel<<<cdiv(E * 32, 256), 256, 0, stream>>>(y2, 48, nrm, row, col, tmp, 20, E);
  add_root_relu2_kernel<<<cdiv(N * 20, 256), 256, 0, stream>>>(tmp, y2, c2_bias, g1, N * 20);

  // t=1: 10x10 per-stack GEMM, propagate, mean+relu+log_softmax
  mid_gemm_w2_kernel<<<cdiv(N * 20, 256), 256, 0, stream>>>(g1, c2_w, z2, N * 20);
  zero_f32x4_kernel<<<cdiv(N * 5, 256), 256, 0, stream>>>((float4*)tmp, N * 5);
  prop_scatter_kernel<<<cdiv(E * 32, 256), 256, 0, stream>>>(z2, 20, nrm, row, col, tmp, 20, E);
  final_out_kernel<<<cdiv(N, 256), 256, 0, stream>>>(tmp, y2, c2_bias, out, N);
}
